// LoRAMoeLayer_56513179681218
// MI455X (gfx1250) — matrix operand order repaired and checked by reading, not compiled, not run on hardware
//
#include <hip/hip_runtime.h>
#include <hip/hip_bf16.h>

// ---------------------------------------------------------------------------
// LoRA-MoE fused layer for MI455X (gfx1250, wave32, WMMA, async-to-LDS).
//
//   out = x @ base_w^T + base_b + SCALING * sum_e w[m,e] * (x @ A_e^T) @ B_e^T
//
// Pipeline:
//   1. gate_kernel      : top-2 softmax routing weights  -> wfull[M,8]
//   2. lora_low_kernel  : u[m,e*R+r] = SCALING*w[m,e]*dot(x[m],A_e[r])
//   3. pack_a_kernel    : bf16x3 split of [x | u]   -> Ahi/Alo [M][4224]
//   4. pack_bt_kernel   : bf16x3 split + TRANSPOSE of base_w -> BThi/BTlo[k][n]
//   5. pack_btl_kernel  : same for lora_B concat rows (k = 4096..4223)
//   6. gemm_kernel      : K=4224 GEMM, double-buffered LDS fed by
//                         global_load_async_to_lds_b128 (ASYNCcnt),
//                         3x v_wmma_f32_16x16x32_bf16 per 16x16x32 tile.
//                         Wave tile 64x64 -> 48 WMMA : 16 b128 A + 128 scalar B ds_loads per step.
//
// Precision: bf16x3 (hi*hi + hi*lo + lo*hi) ~ f32-class accuracy at bf16
// XDL rate; conversions are done ONCE in pack kernels, not per block tile.
// Workspace: ~141 MB (wfull 128K + u 2M + 4 x 34.6M bf16 panels).
// ---------------------------------------------------------------------------

typedef __attribute__((ext_vector_type(16))) __bf16 v16bf;
typedef __attribute__((ext_vector_type(8)))  __bf16 v8bf;
typedef __attribute__((ext_vector_type(4)))  __bf16 v4bf;
typedef __attribute__((ext_vector_type(8)))  float  v8f;

#define MTOK   4096           // B*S tokens
#define DDIM   4096           // D_IN
#define ODIM   4096           // D_OUT
#define NEXP   8
#define RLORA  16
#define ERDIM  (NEXP*RLORA)   // 128
#define KTOT   (DDIM + ERDIM) // 4224
#define SCALING 2.0f

__device__ __forceinline__ void splitf(float x, __bf16 &h, __bf16 &l) {
    h = (__bf16)x;
    l = (__bf16)(x - (float)h);
}

// CDNA5 async global->LDS copy, 16B per lane, tracked by ASYNCcnt.
__device__ __forceinline__ void async_b128(const void* lds_dst, const void* gsrc) {
    asm volatile("global_load_async_to_lds_b128 %0, %1, off"
                 :: "v"((unsigned)(size_t)lds_dst),
                    "v"((unsigned long long)(size_t)gsrc)
                 : "memory");
}

// ---------------------------------------------------------------------------
// Kernel 1: gate logits -> top-2 softmax -> dense per-expert weight map.
// One wave per token; lanes stripe over D.
// ---------------------------------------------------------------------------
__global__ __launch_bounds__(256) void gate_kernel(
        const float* __restrict__ x, const float* __restrict__ gw,
        float* __restrict__ wfull)
{
    const int lane  = threadIdx.x & 31;
    const int wid   = threadIdx.x >> 5;
    const int token = blockIdx.x * 8 + wid;

    float p[NEXP];
#pragma unroll
    for (int e = 0; e < NEXP; ++e) p[e] = 0.f;

    const float* xr = x + (size_t)token * DDIM;
    for (int d = lane * 4; d < DDIM; d += 32 * 4) {
        float4 xv = *(const float4*)(xr + d);
#pragma unroll
        for (int e = 0; e < NEXP; ++e) {
            float4 g = *(const float4*)(gw + (size_t)e * DDIM + d);
            p[e] += xv.x * g.x + xv.y * g.y + xv.z * g.z + xv.w * g.w;
        }
    }
#pragma unroll
    for (int e = 0; e < NEXP; ++e)
        for (int off = 16; off > 0; off >>= 1)
            p[e] += __shfl_xor(p[e], off, 32);

    if (lane == 0) {
        int   b1 = 0;  float v1 = p[0];
#pragma unroll
        for (int e = 1; e < NEXP; ++e) if (p[e] > v1) { v1 = p[e]; b1 = e; }
        int   b2 = -1; float v2 = -3.4e38f;
#pragma unroll
        for (int e = 0; e < NEXP; ++e) if (e != b1 && p[e] > v2) { v2 = p[e]; b2 = e; }
        float e2 = __expf(v2 - v1);
        float s  = 1.f + e2;
        float w1 = 1.f / s, w2 = e2 / s;
        float* wr = wfull + (size_t)token * NEXP;
#pragma unroll
        for (int e = 0; e < NEXP; ++e)
            wr[e] = (e == b1) ? w1 : ((e == b2) ? w2 : 0.f);
    }
}

// ---------------------------------------------------------------------------
// Kernel 2: u[m, e*R+r] = SCALING * w[m,e] * dot(x[m], lora_A[e,r,:])
// 4 tokens per block; x staged in LDS; lora_A (2MB) stays L2-resident.
// ---------------------------------------------------------------------------
#define UTOK 4
__global__ __launch_bounds__(128) void lora_low_kernel(
        const float* __restrict__ x, const float* __restrict__ lA,
        const float* __restrict__ wfull, float* __restrict__ u)
{
    __shared__ float xs[UTOK][1024];
    const int t  = threadIdx.x;              // 0..127 = (e*16 + r)
    const int m0 = blockIdx.x * UTOK;

    float acc[UTOK];
#pragma unroll
    for (int tk = 0; tk < UTOK; ++tk) acc[tk] = 0.f;

    for (int c = 0; c < DDIM; c += 1024) {
        __syncthreads();
#pragma unroll
        for (int tk = 0; tk < UTOK; ++tk) {
            const float4* src = (const float4*)(x + (size_t)(m0 + tk) * DDIM + c + t * 8);
            *(float4*)&xs[tk][t * 8]     = src[0];
            *(float4*)&xs[tk][t * 8 + 4] = src[1];
        }
        __syncthreads();
        const float* ar = lA + (size_t)t * DDIM + c;
        for (int i = 0; i < 1024; i += 4) {
            float4 av = *(const float4*)(ar + i);
#pragma unroll
            for (int tk = 0; tk < UTOK; ++tk) {
                float4 xv = *(const float4*)&xs[tk][i];
                acc[tk] += av.x * xv.x + av.y * xv.y + av.z * xv.z + av.w * xv.w;
            }
        }
    }
    const int e = t >> 4;
#pragma unroll
    for (int tk = 0; tk < UTOK; ++tk)
        u[(size_t)(m0 + tk) * ERDIM + t] =
            SCALING * wfull[(size_t)(m0 + tk) * NEXP + e] * acc[tk];
}

// ---------------------------------------------------------------------------
// Kernel 3: pack A panel = [x | u] split to bf16 hi/lo, row stride KTOT.
// ---------------------------------------------------------------------------
__global__ __launch_bounds__(256) void pack_a_kernel(
        const float* __restrict__ x, const float* __restrict__ u,
        __bf16* __restrict__ Ah, __bf16* __restrict__ Al)
{
    const size_t g  = (size_t)blockIdx.x * 256 + threadIdx.x; // group of 4
    const int row = (int)(g / (KTOT / 4));
    const int c4  = (int)(g % (KTOT / 4)) * 4;

    float4 f = (c4 < DDIM)
        ? *(const float4*)(x + (size_t)row * DDIM + c4)
        : *(const float4*)(u + (size_t)row * ERDIM + (c4 - DDIM));

    union { v4bf v; __bf16 e[4]; } hh, ll;
    splitf(f.x, hh.e[0], ll.e[0]);  splitf(f.y, hh.e[1], ll.e[1]);
    splitf(f.z, hh.e[2], ll.e[2]);  splitf(f.w, hh.e[3], ll.e[3]);
    *(v4bf*)(Ah + (size_t)row * KTOT + c4) = hh.v;
    *(v4bf*)(Al + (size_t)row * KTOT + c4) = ll.v;
}

// ---------------------------------------------------------------------------
// Kernel 4: pack + transpose base_w[n][k] -> BT[k][n] bf16 hi/lo (64x64 tiles).
// ---------------------------------------------------------------------------
__global__ __launch_bounds__(256) void pack_bt_kernel(
        const float* __restrict__ bw,
        __bf16* __restrict__ BTh, __bf16* __restrict__ BTl)
{
    __shared__ float ts[64][68];
    const int t  = threadIdx.x;
    const int kb = blockIdx.x * 64;
    const int nb = blockIdx.y * 64;

#pragma unroll
    for (int i = 0; i < 4; ++i) {
        const int n_l = (t >> 4) + i * 16;
        const int k_l = (t & 15) * 4;
        float4 f = *(const float4*)(bw + (size_t)(nb + n_l) * DDIM + kb + k_l);
        *(float4*)&ts[n_l][k_l] = f;
    }
    __syncthreads();
#pragma unroll
    for (int i = 0; i < 4; ++i) {
        const int k_l = (t >> 4) + i * 16;
        const int n_l = (t & 15) * 4;
        union { v4bf v; __bf16 e[4]; } hh, ll;
#pragma unroll
        for (int j = 0; j < 4; ++j)
            splitf(ts[n_l + j][k_l], hh.e[j], ll.e[j]);
        *(v4bf*)(BTh + (size_t)(kb + k_l) * ODIM + nb + n_l) = hh.v;
        *(v4bf*)(BTl + (size_t)(kb + k_l) * ODIM + nb + n_l) = ll.v;
    }
}

// ---------------------------------------------------------------------------
// Kernel 5: pack lora_B concat rows: BT[4096 + e*16 + r][n] = lora_B[e][n][r].
// ---------------------------------------------------------------------------
__global__ __launch_bounds__(256) void pack_btl_kernel(
        const float* __restrict__ lB,
        __bf16* __restrict__ BTh, __bf16* __restrict__ BTl)
{
    const int idx = blockIdx.x * 256 + threadIdx.x;
    const int er  = idx >> 12;        // 0..127
    const int n   = idx & 4095;
    const int e   = er >> 4, r = er & 15;
    float f = lB[((size_t)e * ODIM + n) * RLORA + r];
    __bf16 h, l;
    splitf(f, h, l);
    BTh[(size_t)(DDIM + er) * ODIM + n] = h;
    BTl[(size_t)(DDIM + er) * ODIM + n] = l;
}

// ---------------------------------------------------------------------------
// Kernel 6: GEMM, K = 4224, bf16x3, double-buffered async-to-LDS pipeline.
// Block tile 128(M) x 256(N), 8 waves (2 M x 4 N), wave tile 64x64
// (4x4 WMMA tiles -> 48 v_wmma per k-step per wave vs 16 ds_load_b128 for A + 128 scalar bf16 B loads).
// ---------------------------------------------------------------------------
#define KSTEP 32
#define NSTEP (KTOT / KSTEP)   // 132
#define APAD  40               // A LDS row stride (bf16): 80B, conflict-free
#define BPAD  264              // B LDS row stride (bf16): 528B, conflict-free

union Frag16 { v16bf f; v8bf h[2]; };

__global__ __launch_bounds__(256) void gemm_kernel(
        const __bf16* __restrict__ Ah, const __bf16* __restrict__ Al,
        const __bf16* __restrict__ BTh, const __bf16* __restrict__ BTl,
        const float* __restrict__ bias, float* __restrict__ out)
{
    __shared__ __bf16 sAh[2][128][APAD];    // 20.0 KB
    __shared__ __bf16 sAl[2][128][APAD];    // 20.0 KB
    __shared__ __bf16 sBh[2][KSTEP][BPAD];  // 33.0 KB
    __shared__ __bf16 sBl[2][KSTEP][BPAD];  // 33.0 KB  -> 106 KB total

    const int t    = threadIdx.x;
    const int lane = t & 31;
    const int wid  = t >> 5;
    const int wm   = (wid & 1) * 64;    // wave M offset (2 waves along M)
    const int wn   = (wid >> 1) * 64;   // wave N offset (4 waves along N)
    const int mb   = blockIdx.y * 128;
    const int nb   = blockIdx.x * 256;

    v8f acc[4][4];
    const v8f vz = {};
#pragma unroll
    for (int mt = 0; mt < 4; ++mt)
#pragma unroll
        for (int nt = 0; nt < 4; ++nt) acc[mt][nt] = vz;

    // Per-thread async copy assignment (12 x b128 issues per stage):
    //   A (128x32 bf16, 8KB/array): thread t -> row t>>1, 32B half (t&1).
    //   B (32x256 bf16, 16KB/array): thread t -> k-row t>>3, 64B seg (t&7).
    const int ar = t >> 1, ac = (t & 1) * 16;   // A row / col (bf16 elems)
    const int br = t >> 3, bc = (t & 7) * 32;   // B k-row / n-col

    auto issue_stage = [&](int ks, int buf) {
        const int k0 = ks * KSTEP;
        const __bf16* gah = Ah + (size_t)(mb + ar) * KTOT + k0 + ac;
        const __bf16* gal = Al + (size_t)(mb + ar) * KTOT + k0 + ac;
        async_b128(&sAh[buf][ar][ac],     gah);
        async_b128(&sAh[buf][ar][ac + 8], gah + 8);
        async_b128(&sAl[buf][ar][ac],     gal);
        async_b128(&sAl[buf][ar][ac + 8], gal + 8);
        const __bf16* gbh = BTh + (size_t)(k0 + br) * ODIM + nb + bc;
        const __bf16* gbl = BTl + (size_t)(k0 + br) * ODIM + nb + bc;
#pragma unroll
        for (int s = 0; s < 4; ++s) {
            async_b128(&sBh[buf][br][bc + s * 8], gbh + s * 8);
            async_b128(&sBl[buf][br][bc + s * 8], gbl + s * 8);
        }
    };

    issue_stage(0, 0);

    const int acb = (lane >> 4) * 8;      // A frag K sub-base per ISA layout

    for (int ks = 0; ks < NSTEP; ++ks) {
        const int buf = ks & 1;
        if (ks + 1 < NSTEP) {
            issue_stage(ks + 1, buf ^ 1);
            // current stage complete when only the 12 next-stage issues remain
            asm volatile("s_wait_asynccnt 0xc" ::: "memory");
        } else {
            asm volatile("s_wait_asynccnt 0x0" ::: "memory");
        }
        __syncthreads();   // all waves' portions of stage ks are in LDS

        // ---- A fragments (ISA 7.12.2): lanes 0-15 row M=lane, K {0..7,16..23};
        //      lanes 16-31 row M=lane-16, K {8..15,24..31}.
        Frag16 ah[4], al[4];
#pragma unroll
        for (int mt = 0; mt < 4; ++mt) {
            const int row = wm + mt * 16 + (lane & 15);
            ah[mt].h[0] = *(const v8bf*)&sAh[buf][row][acb];
            ah[mt].h[1] = *(const v8bf*)&sAh[buf][row][acb + 16];
            al[mt].h[0] = *(const v8bf*)&sAl[buf][row][acb];
            al[mt].h[1] = *(const v8bf*)&sAl[buf][row][acb + 16];
        }
        // ---- B fragments per n-tile (lane holds column N=lane&15, K same as A frag);
        //      loaded inside the nt loop to cap register pressure.
#pragma unroll
        for (int nt = 0; nt < 4; ++nt) {
            Frag16 bh, bl;
            const int nc = wn + nt * 16 + (lane & 15);
#pragma unroll
            for (int j = 0; j < 8; ++j) {
                bh.f[j]     = sBh[buf][acb + j][nc];
                bh.f[8 + j] = sBh[buf][acb + 16 + j][nc];
                bl.f[j]     = sBl[buf][acb + j][nc];
                bl.f[8 + j] = sBl[buf][acb + 16 + j][nc];
            }
            // bf16x3: hi*hi + hi*lo + lo*hi
#pragma unroll
            for (int mt = 0; mt < 4; ++mt) {
                acc[mt][nt] = __builtin_amdgcn_wmma_f32_16x16x32_bf16(
                    false, ah[mt].f, false, bh.f, (short)0, acc[mt][nt], false, false);
                acc[mt][nt] = __builtin_amdgcn_wmma_f32_16x16x32_bf16(
                    false, ah[mt].f, false, bl.f, (short)0, acc[mt][nt], false, false);
                acc[mt][nt] = __builtin_amdgcn_wmma_f32_16x16x32_bf16(
                    false, al[mt].f, false, bh.f, (short)0, acc[mt][nt], false, false);
            }
        }

        __syncthreads();   // buf free for reuse before next-next issue
    }

    // ---- epilogue: +bias, store. C layout: lane<16 -> N=lane, M=vgpr;
    //      lane>=16 -> N=lane-16, M=vgpr+8.
    const int mofs = (lane >> 4) * 8;
#pragma unroll
    for (int nt = 0; nt < 4; ++nt) {
        const int n  = nb + wn + nt * 16 + (lane & 15);
        const float bv = bias[n];
#pragma unroll
        for (int mt = 0; mt < 4; ++mt) {
#pragma unroll
            for (int v = 0; v < 8; ++v) {
                const int m = mb + wm + mt * 16 + v + mofs;
                out[(size_t)m * ODIM + n] = acc[mt][nt][v] + bv;
            }
        }
    }
}

// ---------------------------------------------------------------------------
extern "C" void kernel_launch(void* const* d_in, const int* in_sizes, int n_in,
                              void* d_out, int out_size, void* d_ws, size_t ws_size,
                              hipStream_t stream)
{
    (void)in_sizes; (void)n_in; (void)out_size; (void)ws_size;
    const float* x      = (const float*)d_in[0];
    const float* gate_w = (const float*)d_in[1];
    const float* base_w = (const float*)d_in[2];
    const float* base_b = (const float*)d_in[3];
    const float* lora_A = (const float*)d_in[4];
    const float* lora_B = (const float*)d_in[5];
    float* out = (float*)d_out;

    // ---- workspace layout (assumes ws >= ~141 MB) -------------------------
    char* p = (char*)d_ws;
    float*  wfull = (float*)p;  p += (size_t)MTOK * NEXP  * sizeof(float);  // 128 KB
    float*  u     = (float*)p;  p += (size_t)MTOK * ERDIM * sizeof(float);  // 2 MB
    __bf16* Ahf   = (__bf16*)p; p += (size_t)MTOK * KTOT * sizeof(__bf16);  // 34.6 MB
    __bf16* Alf   = (__bf16*)p; p += (size_t)MTOK * KTOT * sizeof(__bf16);
    __bf16* BTh   = (__bf16*)p; p += (size_t)KTOT * ODIM * sizeof(__bf16);
    __bf16* BTl   = (__bf16*)p; p += (size_t)KTOT * ODIM * sizeof(__bf16);

    gate_kernel    <<<MTOK / 8,    256, 0, stream>>>(x, gate_w, wfull);
    lora_low_kernel<<<MTOK / UTOK, 128, 0, stream>>>(x, lora_A, wfull, u);
    pack_a_kernel  <<<(MTOK * (KTOT / 4)) / 256, 256, 0, stream>>>(x, u, Ahf, Alf);
    pack_bt_kernel <<<dim3(DDIM / 64, ODIM / 64), 256, 0, stream>>>(base_w, BTh, BTl);
    pack_btl_kernel<<<(ERDIM * ODIM) / 256, 256, 0, stream>>>(lora_B, BTh, BTl);
    gemm_kernel    <<<dim3(ODIM / 256, MTOK / 128), 256, 0, stream>>>(
        Ahf, Alf, BTh, BTl, base_b, out);
}
